// BaseAttentionModule_35493609734830
// MI455X (gfx1250) — matrix operand order repaired and checked by reading, not compiled, not run on hardware
//
#include <hip/hip_runtime.h>

// ---------------------------------------------------------------------------
// MI455X (gfx1250) fused causal attention-score kernel.
//   scores = softmax( causal_mask( (X @ W) @ X^T / sqrt(D) ) )
// B=4, S=2048, D=1024. bf16 WMMA (v_wmma_f32_16x16x32_bf16), f32 accumulate.
// v4: ping-pong (copy-free) double buffering in the XW GEMM, and TDM
//     tensor_load_to_lds staging of the shared A block in the scores kernel
//     (TENSORcnt-tracked DMA). Uses the 6-arg clang-23 builtin form.
// ---------------------------------------------------------------------------

typedef __bf16 bf16;
typedef __attribute__((ext_vector_type(16))) __bf16        v16bf;
typedef __attribute__((ext_vector_type(8)))  float         v8f;
typedef __attribute__((ext_vector_type(4)))  unsigned int  v4u;
typedef __attribute__((ext_vector_type(4)))  unsigned int  u32x4;
typedef __attribute__((ext_vector_type(8)))  int           i32x8;
typedef __attribute__((ext_vector_type(4)))  int           i32x4;

union FragU { v16bf f; v4u u[2]; };

// A-fragment (16x32 bf16, row-major source, row stride ld):
// lanes 0-15: row=lane,  K = [0..7] and [16..23]
// lanes16-31: row=lane-16, K = [8..15] and [24..31]
__device__ __forceinline__ v16bf load_a_frag(const bf16* __restrict__ base,
                                             int ld, int k0) {
  const int lane = threadIdx.x & 31;
  const int r    = lane & 15;
  const int kb   = (lane >> 4) << 3;          // 0 or 8
  const bf16* p  = base + (size_t)r * ld + k0 + kb;
  FragU fr;
  fr.u[0] = *(const v4u*)(p);                 // K kb..kb+7
  fr.u[1] = *(const v4u*)(p + 16);            // K kb+16..kb+23
  return fr.f;
}

// B-fragment (32x16 bf16). Source stored column-major: row n of `base` is
// column n of B (i.e. a row of X, or a row of W^T).
// lanes 0-15: col=lane,  K = [0..7] and [16..23];  lanes 16-31: col=lane-16, K = [8..15] and [24..31]
__device__ __forceinline__ v16bf load_b_frag(const bf16* __restrict__ base,
                                             int ld, int k0) {
  const int lane = threadIdx.x & 31;
  const int n    = lane & 15;
  const int kb   = (lane >> 4) << 3;          // 0 or 8
  const bf16* p  = base + (size_t)n * ld + k0 + kb;
  FragU fr;
  fr.u[0] = *(const v4u*)(p);                 // K kb..kb+7
  fr.u[1] = *(const v4u*)(p + 16);            // K kb+16..kb+23
  return fr.f;
}

__device__ __forceinline__ v8f wmma_bf16(v16bf a, v16bf b, v8f c) {
  return __builtin_amdgcn_wmma_f32_16x16x32_bf16(false, a, false, b,
                                                 (short)0, c, false, false);
}

// --------------------------- conversion kernels ----------------------------
__global__ void k_cvt_x(const float* __restrict__ x, bf16* __restrict__ xb,
                        int n) {
  int i = (blockIdx.x * blockDim.x + threadIdx.x) * 4;
  if (i + 3 < n) {
    float4 v = *(const float4*)(x + i);
    xb[i + 0] = (bf16)v.x;
    xb[i + 1] = (bf16)v.y;
    xb[i + 2] = (bf16)v.z;
    xb[i + 3] = (bf16)v.w;
  }
}

// W[k][n] -> Wt[n][k] (bf16), so WMMA B-fragments read contiguous memory.
__global__ void k_cvt_w(const float* __restrict__ w, bf16* __restrict__ wt) {
  const int D = 1024;
  int tid = blockIdx.x * blockDim.x + threadIdx.x;   // linear over W (coalesced read)
  int k = tid >> 10;
  int n = tid & 1023;
  wt[(size_t)n * D + k] = (bf16)w[tid];
}

// ------------------------------ XW = X @ W ---------------------------------
// One wave computes a 16x64 tile of XW (4 accumulators, shared A-fragment).
// Ping-pong double buffering, k unrolled by 64: while WMMAs consume buffer A,
// buffer B's fragments are in flight (and vice versa) -- no register copies,
// WMMA waits are partial loadcnt waits.
__global__ __launch_bounds__(256) void
k_gemm_xw(const bf16* __restrict__ xb, const bf16* __restrict__ wt,
          bf16* __restrict__ xwb) {
  const int K = 1024, LD = 1024;
  const int wave  = (blockIdx.x * blockDim.x + threadIdx.x) >> 5;
  const int mtile = wave >> 4;          // 0..511
  const int nb    = wave & 15;          // 0..15 -> 64-wide N chunk
  const int m0    = mtile << 4;
  const int n0    = nb << 6;
  const bf16* abase = xb + (size_t)m0 * LD;

  v8f acc[4] = {};
  v16bf aA, bA[4], aB, bB[4];

  aA = load_a_frag(abase, LD, 0);
#pragma unroll
  for (int t = 0; t < 4; ++t)
    bA[t] = load_b_frag(wt + (size_t)(n0 + t * 16) * LD, LD, 0);

  for (int k0 = 0; k0 < K; k0 += 64) {
    // load k0+32 into B buffers; compute k0 from A buffers
    aB = load_a_frag(abase, LD, k0 + 32);
#pragma unroll
    for (int t = 0; t < 4; ++t)
      bB[t] = load_b_frag(wt + (size_t)(n0 + t * 16) * LD, LD, k0 + 32);
#pragma unroll
    for (int t = 0; t < 4; ++t)
      acc[t] = wmma_bf16(aA, bA[t], acc[t]);

    // load k0+64 into A buffers (except last trip); compute k0+32 from B
    if (k0 + 64 < K) {
      aA = load_a_frag(abase, LD, k0 + 64);
#pragma unroll
      for (int t = 0; t < 4; ++t)
        bA[t] = load_b_frag(wt + (size_t)(n0 + t * 16) * LD, LD, k0 + 64);
    }
#pragma unroll
    for (int t = 0; t < 4; ++t)
      acc[t] = wmma_bf16(aB, bB[t], acc[t]);
  }

  const int lane = threadIdx.x & 31;
  const int col  = lane & 15;
  const int rofs = (lane >> 4) << 3;    // 0 or 8
#pragma unroll
  for (int t = 0; t < 4; ++t) {
    int n = n0 + t * 16 + col;
#pragma unroll
    for (int r = 0; r < 8; ++r) {
      int m = m0 + rofs + r;
      xwb[(size_t)m * LD + n] = (bf16)acc[t][r];
    }
  }
}

// --------------- fused scores = XW @ X^T, causal mask, softmax -------------
// One block (512 thr = 16 waves) owns 16 rows of one batch.
// LDS: 32 KB staged A block (TDM DMA, shared by all 16 waves, read via
//      ds_load_b128) + 128 KB f32 score block = 160 KB -> 2 blocks per WGP.
// Causal skip: only col tiles ct <= rb are computed; the rest output 0.
__global__ __launch_bounds__(512) void
k_scores_softmax(const bf16* __restrict__ xwb, const bf16* __restrict__ xb,
                 float* __restrict__ out) {
  extern __shared__ char smem[];
  bf16*  As = (bf16*)smem;                      // [16][1024]  (32 KB)
  float* sc = (float*)(smem + 16 * 1024 * sizeof(bf16));  // [16][2048] (128 KB)

  const int S = 2048, D = 1024;
  const int b    = blockIdx.x >> 7;             // batch
  const int rb   = blockIdx.x & 127;            // row block in batch
  const int row0 = rb << 4;
  const bf16* A = xwb + ((size_t)b * S + row0) * D;   // 16 rows of XW (contiguous)
  const bf16* X = xb + (size_t)b * S * D;             // keys (columns of B)

  const int tid  = threadIdx.x;
  const int wave = tid >> 5;                    // 0..15
  const int lane = tid & 31;
  const int col  = lane & 15;
  const int rofs = (lane >> 4) << 3;
  const int ntiles = rb + 1;                    // causal: tiles 0..rb needed
  const float scale = 0.03125f;                 // 1/sqrt(1024)

  // Phase 0: stage the 16xD A block (32 KB, contiguous) into LDS via the
  // Tensor Data Mover: one wave issues the DMA, waits on TENSORcnt, and the
  // workgroup barrier releases everyone else.
#if __has_builtin(__builtin_amdgcn_tensor_load_to_lds) && \
    __has_builtin(__builtin_amdgcn_s_wait_tensorcnt)
  if (wave == 0) {
    // TDM D# descriptor: 1 x 16384 tile of 2-byte elements (fully linear).
    // LDS raw address = low 32 bits of the generic pointer (LDS aperture).
    const unsigned lds_addr = (unsigned)(unsigned long long)(const void*)As;
    const unsigned long long ga = (unsigned long long)(const void*)A;
    u32x4 g0;
    g0[0] = 1u;                                   // count=1, user mode
    g0[1] = lds_addr;                             // lds_addr
    g0[2] = (unsigned)(ga & 0xFFFFFFFFull);       // global_addr[31:0]
    g0[3] = (unsigned)((ga >> 32) & 0x01FFFFFFu)  // global_addr[56:32]
          | (2u << 30);                           // type=2 ("image")
    i32x8 g1;
    g1[0] = 0x00010000;            // workgroup_mask=0, data_size=1 (2 bytes)
    g1[1] = (int)(0x4000u << 16);  // tensor_dim0 = 16384 (bits 79:48, low half)
    g1[2] = (int)(1u << 16);       // tensor_dim0 hi=0 | tensor_dim1 = 1
    g1[3] = (int)(0x4000u << 16);  // tensor_dim1 hi=0 | tile_dim0 = 16384
    g1[4] = 1;                     // tile_dim1 = 1, tile_dim2 = 0
    g1[5] = 16384;                 // tensor_dim0_stride (low 32)
    g1[6] = 0;                     // stride0 hi | tensor_dim1_stride lo
    g1[7] = 0;
    i32x4 z4 = {};
    i32x8 z8 = {};
    __builtin_amdgcn_tensor_load_to_lds(g0, g1, z4, z4, z8, 0);
    __builtin_amdgcn_s_wait_tensorcnt(0);
  }
#else
  {
    const uint4* src = (const uint4*)A;         // 2048 16-byte quads
    uint4* dst = (uint4*)As;
#pragma unroll
    for (int i = 0; i < 4; ++i) dst[tid + i * 512] = src[tid + i * 512];
  }
#endif
  __syncthreads();

  // Phase 1: per-wave col tiles ct = wave + 16*i (wave-uniform trip counts,
  // EXEC stays all-ones around the WMMAs). K-outer; A frag from LDS once per
  // k-step; B frags in groups of 4 so WMMA waits are partial loadcnt waits.
  v8f acc[8] = {};
  int mytiles = 0;
  for (int ct = wave; ct < ntiles; ct += 16) ++mytiles;

  if (mytiles > 0) {
    for (int k0 = 0; k0 < D; k0 += 32) {
      v16bf a = load_a_frag(As, D, k0);         // ds_load_b128 x2 (DScnt)
      // prefetch the wave's first B tile 8 k-steps ahead (gfx1250 prefetch)
      __builtin_prefetch(X + (size_t)(wave << 4) * D + k0 + 256, 0, 0);
#pragma unroll
      for (int g = 0; g < 8; g += 4) {
        v16bf bfr[4];
#pragma unroll
        for (int u = 0; u < 4; ++u) {
          int i = g + u;
          if (i < mytiles) {
            int j0 = (wave + (i << 4)) << 4;
            bfr[u] = load_b_frag(X + (size_t)j0 * D, D, k0);
          }
        }
#pragma unroll
        for (int u = 0; u < 4; ++u) {
          int i = g + u;
          if (i < mytiles) acc[i] = wmma_bf16(a, bfr[u], acc[i]);
        }
      }
    }
#pragma unroll 8
    for (int i = 0; i < 8; ++i) {
      if (i < mytiles) {
        int j = ((wave + (i << 4)) << 4) + col;
#pragma unroll
        for (int r = 0; r < 8; ++r) {
          int rr = rofs + r;                    // local row
          int gi = row0 + rr;                   // global row
          float v = (j > gi) ? -__builtin_inff() : acc[i][r] * scale;
          sc[rr * S + j] = v;
        }
      }
    }
  }
  __syncthreads();

  // Phase 2: softmax. Wave w handles row w over the jlimit valid columns.
  {
    const int row    = wave;
    const int jlimit = ntiles << 4;
    float* rowp = sc + row * S;
    float m = -__builtin_inff();
    for (int j = lane; j < jlimit; j += 32) m = fmaxf(m, rowp[j]);
#pragma unroll
    for (int off = 16; off >= 1; off >>= 1) m = fmaxf(m, __shfl_xor(m, off, 32));
    float s = 0.f;
    for (int j = lane; j < jlimit; j += 32) {
      float e = __expf(rowp[j] - m);
      rowp[j] = e;
      s += e;
    }
#pragma unroll
    for (int off = 16; off >= 1; off >>= 1) s += __shfl_xor(s, off, 32);
    const float inv = 1.0f / s;
    float* op = out + ((size_t)b * S + (row0 + row)) * S;
    for (int j = lane; j < jlimit; j += 32) op[j] = rowp[j] * inv;
    for (int j = jlimit + lane; j < S; j += 32) op[j] = 0.f;  // fully masked
  }
}

// ------------------------------- launcher ----------------------------------
extern "C" void kernel_launch(void* const* d_in, const int* in_sizes, int n_in,
                              void* d_out, int out_size, void* d_ws,
                              size_t ws_size, hipStream_t stream) {
  const float* x = (const float*)d_in[0];        // [4,2048,1024]
  const float* w = (const float*)d_in[1];        // [1024,1024]
  float* out = (float*)d_out;                    // [4,2048,2048]

  const int B = 4, S = 2048, D = 1024;
  const size_t nx = (size_t)B * S * D;           // 8,388,608

  // workspace: x_bf16 (16 MB) | Wt_bf16 (2 MB) | xw_bf16 (16 MB) = 34 MB
  bf16* xb  = (bf16*)d_ws;
  bf16* wt  = xb + nx;
  bf16* xwb = wt + (size_t)D * D;

  k_cvt_x<<<(int)(nx / (256 * 4)), 256, 0, stream>>>(x, xb, (int)nx);
  k_cvt_w<<<(D * D) / 256, 256, 0, stream>>>(w, wt);
  k_gemm_xw<<<1024, 256, 0, stream>>>(xb, wt, xwb);          // 8192 wave-tiles
  const int lds_bytes = 16 * 1024 * (int)sizeof(bf16)        // A stage: 32 KB
                      + 16 * S * (int)sizeof(float);         // scores: 128 KB
  k_scores_softmax<<<B * (S / 16), 512, lds_bytes, stream>>>(xwb, xb, out);
}
